// MSAModel_26396869001464
// MI455X (gfx1250) — matrix operand order repaired and checked by reading, not compiled, not run on hardware
//
#include <hip/hip_runtime.h>

#define BB      8
#define NN      2048
#define DD      128
#define LAYERSN 3

typedef __attribute__((ext_vector_type(16))) __bf16 v16bf;
typedef __attribute__((ext_vector_type(8)))  float  v8f;

__device__ __forceinline__ v8f wmma_bf16(v16bf a, v16bf b, v8f c) {
    // 8 args: (neg_a, A, neg_b, B, c_mod, C, reuse_a, reuse_b)
    return __builtin_amdgcn_wmma_f32_16x16x32_bf16(
        false, a, false, b, (short)0, c, false, false);
}

// Load 16 f32 (two 32B chunks at p and p+8) and convert to a bf16 A-fragment.
__device__ __forceinline__ v16bf load_cvt_afrag(const float* p) {
    const float4* q = (const float4*)p;
    float4 f0 = q[0], f1 = q[1];   // K chunk c0 .. c0+7
    float4 f2 = q[2], f3 = q[3];   // K chunk c0+8 .. c0+15
    v16bf a;
    a[0]  = (__bf16)f0.x; a[1]  = (__bf16)f0.y;
    a[2]  = (__bf16)f0.z; a[3]  = (__bf16)f0.w;
    a[4]  = (__bf16)f1.x; a[5]  = (__bf16)f1.y;
    a[6]  = (__bf16)f1.z; a[7]  = (__bf16)f1.w;
    a[8]  = (__bf16)f2.x; a[9]  = (__bf16)f2.y;
    a[10] = (__bf16)f2.z; a[11] = (__bf16)f2.w;
    a[12] = (__bf16)f3.x; a[13] = (__bf16)f3.y;
    a[14] = (__bf16)f3.z; a[15] = (__bf16)f3.w;
    return a;
}

// ---------------------------------------------------------------------------
// denomInv[b*N + i] = 1 / (sum_j adj[b,i,j] + 1)    (one wave per row)
// ---------------------------------------------------------------------------
__global__ __launch_bounds__(256) void row_denom(const float* __restrict__ adj,
                                                 float* __restrict__ dinv) {
    int row  = blockIdx.x * 8 + (threadIdx.x >> 5);   // 8 waves per block
    int lane = threadIdx.x & 31;
    const float* r = adj + (size_t)row * NN;
    float s = 0.f;
    for (int j = lane; j < NN; j += 32) s += r[j];
    #pragma unroll
    for (int off = 16; off > 0; off >>= 1) s += __shfl_xor(s, off);
    if (lane == 0) dinv[row] = 1.0f / (s + 1.0f);
}

// ---------------------------------------------------------------------------
// Xt[b][d][n] = (bf16) inputs[b][n][d]
// ---------------------------------------------------------------------------
__global__ __launch_bounds__(256) void prep_x(const float* __restrict__ in,
                                              __bf16* __restrict__ Xt) {
    int idx = blockIdx.x * 256 + threadIdx.x;         // over B*N*D
    int d = idx % DD;
    int n = (idx / DD) % NN;
    int b = idx / (DD * NN);
    Xt[((size_t)b * DD + d) * NN + n] = (__bf16)in[idx];
}

// ---------------------------------------------------------------------------
// Wt[l][o][i] = (bf16) W[l][i][o]
// ---------------------------------------------------------------------------
__global__ __launch_bounds__(256) void prep_w(const float* __restrict__ W,
                                              __bf16* __restrict__ Wt) {
    int idx = blockIdx.x * 256 + threadIdx.x;         // over L*D*D
    int o = idx % DD;
    int i = (idx / DD) % DD;
    int l = idx / (DD * DD);
    Wt[((size_t)l * DD + o) * DD + i] = (__bf16)W[idx];
}

// ---------------------------------------------------------------------------
// Fused GCN layer. Block = 128 threads = 4 waves; block owns 128 rows x D=128.
// Each wave owns 32 rows (two 16-row halves sharing every B fragment, so each
// B-tile load feeds two WMMAs). No inter-wave cooperation -> no barriers.
// ---------------------------------------------------------------------------
__global__ __launch_bounds__(128)
void gcn_layer(const float*  __restrict__ adj,
               const __bf16* __restrict__ Xt,     // [B][D][N] bf16
               const __bf16* __restrict__ Wt,     // [D_out][D_in] bf16 (layer slice)
               const float*  __restrict__ bias,   // [D] f32 (layer slice)
               const float*  __restrict__ dinv,   // [B*N] f32
               __bf16*       __restrict__ XtOut,  // [B][D][N] bf16, or null
               float*        __restrict__ fOut,   // [B][N][D] f32, or null
               int last)
{
    // Per-wave staging of the 32x128 Ax strip, bf16, padded stride 136
    // (272B row stride, multiple of 16B -> aligned b128 chunk reads).
    __shared__ __align__(16) __bf16 axLds[4][32][136];

    const int lane  = threadIdx.x & 31;
    const int wave  = threadIdx.x >> 5;
    const int l16   = lane & 15;
    const int hhi   = lane >> 4;                 // half-wave select
    const int batch = blockIdx.x >> 4;
    const int rowBase = (blockIdx.x & 15) * 128 + wave * 32;

    // A-fragment K chunks (contiguous, matching B): lanes 0-15 hold K{0..15},
    // lanes 16-31 hold K{16..31}.
    const int c0  = hhi * 16;   // A chunk base inside 32-wide K tile
    const int klo = hhi * 16;   // B chunk base inside 32-wide K tile

    const float*  aRow0 = adj + ((size_t)batch * NN + rowBase + l16) * NN;
    const float*  aRow1 = aRow0 + (size_t)16 * NN;
    const __bf16* xBase = Xt + (size_t)batch * DD * NN;

    v8f acc[2][8];
    #pragma unroll
    for (int h = 0; h < 2; ++h)
        #pragma unroll
        for (int g = 0; g < 8; ++g) acc[h][g] = {};

    // ----------------- GEMM1: Ax = adj @ X, K = 0..2047 step 32 ------------
    for (int kk = 0; kk < NN; kk += 32) {
        // Prefetch the adj stream ~4KB ahead (emits global_prefetch_b8).
        __builtin_prefetch(aRow0 + kk + 1024, 0, 1);
        __builtin_prefetch(aRow1 + kk + 1024, 0, 1);

        v16bf a0 = load_cvt_afrag(aRow0 + kk + c0);
        v16bf a1 = load_cvt_afrag(aRow1 + kk + c0);

        #pragma unroll
        for (int g = 0; g < 8; ++g) {
            const int col = g * 16 + l16;
            v16bf b = *(const v16bf*)(xBase + (size_t)col * NN + kk + klo);
            acc[0][g] = wmma_bf16(a0, b, acc[0][g]);
            acc[1][g] = wmma_bf16(a1, b, acc[1][g]);
        }
    }

    // ---- stage Ax strip into LDS as bf16 (C layout: row = p + 8*hhi) ------
    #pragma unroll
    for (int h = 0; h < 2; ++h)
        #pragma unroll
        for (int g = 0; g < 8; ++g)
            #pragma unroll
            for (int p = 0; p < 8; ++p)
                axLds[wave][16 * h + p + 8 * hhi][g * 16 + l16] =
                    (__bf16)acc[h][g][p];

    // ----------------- GEMM2: Y = Ax @ W, K = 0..127 step 32 ---------------
    v8f acc2[2][8];
    #pragma unroll
    for (int h = 0; h < 2; ++h)
        #pragma unroll
        for (int g = 0; g < 8; ++g) acc2[h][g] = {};

    #pragma unroll
    for (int kk = 0; kk < 4; ++kk) {
        v16bf a2[2];
        #pragma unroll
        for (int h = 0; h < 2; ++h) {
            const __bf16* arow = &axLds[wave][16 * h + l16][kk * 32 + c0];
            union { uint4 u[2]; v16bf v; } t;
            t.u[0] = ((const uint4*)arow)[0];          // K chunk c0..c0+7
            t.u[1] = ((const uint4*)(arow + 8))[0];    // K chunk c0+8..c0+15
            a2[h] = t.v;
        }
        #pragma unroll
        for (int g = 0; g < 8; ++g) {
            const int col = g * 16 + l16;
            v16bf b = *(const v16bf*)(Wt + (size_t)col * DD + kk * 32 + klo);
            acc2[0][g] = wmma_bf16(a2[0], b, acc2[0][g]);
            acc2[1][g] = wmma_bf16(a2[1], b, acc2[1][g]);
        }
    }

    // ----------------- epilogue: bias, *denomInv, relu, store --------------
    #pragma unroll
    for (int h = 0; h < 2; ++h) {
        #pragma unroll
        for (int p = 0; p < 8; ++p) {
            const int row = rowBase + 16 * h + p + 8 * hhi;
            const float di = dinv[batch * NN + row];
            #pragma unroll
            for (int g = 0; g < 8; ++g) {
                const int col = g * 16 + l16;
                float v = (acc2[h][g][p] + bias[col]) * di;
                v = v > 0.f ? v : 0.f;
                if (last) {
                    fOut[((size_t)batch * NN + row) * DD + col] = v;
                } else {
                    XtOut[((size_t)batch * DD + col) * NN + row] = (__bf16)v;
                }
            }
        }
    }
}

// ---------------------------------------------------------------------------
extern "C" void kernel_launch(void* const* d_in, const int* in_sizes, int n_in,
                              void* d_out, int out_size, void* d_ws, size_t ws_size,
                              hipStream_t stream) {
    const float* inputs = (const float*)d_in[0];   // [B,N,D]   f32
    const float* adj    = (const float*)d_in[1];   // [B,N,N]   f32
    const float* W      = (const float*)d_in[2];   // [L,D,D]   f32
    const float* bias   = (const float*)d_in[3];   // [L,D]     f32
    float* out = (float*)d_out;                    // [B,N,D]   f32

    char* ws = (char*)d_ws;
    float*  dinv = (float*)(ws);                                  //  64 KB
    __bf16* Wt   = (__bf16*)(ws + 65536);                         //  96 KB
    __bf16* Xa   = (__bf16*)(ws + 163840);                        //   4 MB
    __bf16* Xb   = (__bf16*)(ws + 163840 + (size_t)BB*DD*NN*2);   //   4 MB

    row_denom<<<BB * NN / 8, 256, 0, stream>>>(adj, dinv);
    prep_x<<<BB * NN * DD / 256, 256, 0, stream>>>(inputs, Xa);
    prep_w<<<LAYERSN * DD * DD / 256, 256, 0, stream>>>(W, Wt);

    const int grid = BB * (NN / 128);  // 128 workgroups, 4 waves each
    gcn_layer<<<grid, 128, 0, stream>>>(adj, Xa, Wt + 0 * DD * DD,
                                        bias + 0 * DD, dinv, Xb, nullptr, 0);
    gcn_layer<<<grid, 128, 0, stream>>>(adj, Xb, Wt + 1 * DD * DD,
                                        bias + 1 * DD, dinv, Xa, nullptr, 0);
    gcn_layer<<<grid, 128, 0, stream>>>(adj, Xa, Wt + 2 * DD * DD,
                                        bias + 2 * DD, dinv, nullptr, out, 1);
}
